// GroupedQueryAttention_17179869184466
// MI455X (gfx1250) — matrix operand order repaired and checked by reading, not compiled, not run on hardware
//
#include <hip/hip_runtime.h>
#include <hip/hip_bf16.h>

#define S_LEN  2048
#define HID_   4096
#define NH_    32
#define NKV_   8
#define HD_    128
#define WIN_   1024

typedef __attribute__((ext_vector_type(16))) __bf16 v16bf;
typedef __attribute__((ext_vector_type(8)))  float  v8f;

__device__ __forceinline__ v8f wmma_bf16(v16bf a, v16bf b, v8f c) {
  // D = A(16x32) * B(32x16) + C, fp32 accumulate
  return __builtin_amdgcn_wmma_f32_16x16x32_bf16(false, a, false, b, (short)0, c,
                                                 false, false);
}

__device__ __forceinline__ v16bf pack16(const float t[16]) {
  v16bf v;
#pragma unroll
  for (int e = 0; e < 16; ++e) v[e] = (__bf16)t[e];
  return v;
}

// A-fragment (16x32 tile, row m = row0 + lane%16, K-contiguous source).
// Per-lane K offsets: {hi*8+0..7} and {16+hi*8+0..7}  (CDNA5 16-bit A layout)
__device__ __forceinline__ v16bf load_a_frag(const float* __restrict__ base, int ld,
                                             int row0, int k0, int lane) {
  const int r = lane & 15, hi = lane >> 4;
  const float* p = base + (size_t)(row0 + r) * ld + k0 + hi * 8;
  float4 a0 = *(const float4*)(p);
  float4 a1 = *(const float4*)(p + 4);
  float4 a2 = *(const float4*)(p + 16);
  float4 a3 = *(const float4*)(p + 20);
  float t[16] = {a0.x, a0.y, a0.z, a0.w, a1.x, a1.y, a1.z, a1.w,
                 a2.x, a2.y, a2.z, a2.w, a3.x, a3.y, a3.z, a3.w};
  return pack16(t);
}

// Same A-fragment gather but from an LDS-resident fp32 tile (ds_load_b128 path).
__device__ __forceinline__ v16bf load_a_frag_lds(const float* p) {
  float4 a0 = *(const float4*)(p);
  float4 a1 = *(const float4*)(p + 4);
  float4 a2 = *(const float4*)(p + 16);
  float4 a3 = *(const float4*)(p + 20);
  float t[16] = {a0.x, a0.y, a0.z, a0.w, a1.x, a1.y, a1.z, a1.w,
                 a2.x, a2.y, a2.z, a2.w, a3.x, a3.y, a3.z, a3.w};
  return pack16(t);
}

// B-fragment (32x16 tile; column n = col0 + lane%16 is a K-contiguous row of the
// row-major source, i.e. B = W^T). Per-lane K offsets: {hi*8+0..7} and {16+hi*8+0..7}.
__device__ __forceinline__ v16bf load_b_frag(const float* __restrict__ base, int ld,
                                             int col0, int k0, int lane) {
  const int n = lane & 15, hi = lane >> 4;
  const float* p = base + (size_t)(col0 + n) * ld + k0 + hi * 8;
  float4 b0 = *(const float4*)(p);
  float4 b1 = *(const float4*)(p + 4);
  float4 b2 = *(const float4*)(p + 16);
  float4 b3 = *(const float4*)(p + 20);
  float t[16] = {b0.x, b0.y, b0.z, b0.w, b1.x, b1.y, b1.z, b1.w,
                 b2.x, b2.y, b2.z, b2.w, b3.x, b3.y, b3.z, b3.w};
  return pack16(t);
}

// Y(MxN) = X(MxK) * W(NxK)^T ; fp32 in/out, bf16 WMMA math, fp32 accumulate.
// Block = 8 waves -> 64x64 tile; wave -> 16x32 tile.
__global__ __launch_bounds__(256) void proj_nt_kernel(const float* __restrict__ X,
                                                      const float* __restrict__ W,
                                                      float* __restrict__ Y,
                                                      int N, int K) {
  const int lane = threadIdx.x & 31;
  const int wave = threadIdx.x >> 5;
  const int wm = blockIdx.x * 64 + (wave >> 1) * 16;
  const int wn = blockIdx.y * 64 + (wave & 1) * 32;
  v8f acc0 = {};
  v8f acc1 = {};
  for (int k0 = 0; k0 < K; k0 += 32) {
    if (k0 + 32 < K) {  // global_prefetch_b8 of next k-slab
      __builtin_prefetch((const void*)(X + (size_t)(wm + (lane & 15)) * K + k0 + 32), 0, 1);
      __builtin_prefetch((const void*)(W + (size_t)(wn + (lane & 15)) * K + k0 + 32), 0, 1);
    }
    v16bf a  = load_a_frag(X, K, wm, k0, lane);
    v16bf b0 = load_b_frag(W, K, wn, k0, lane);
    v16bf b1 = load_b_frag(W, K, wn + 16, k0, lane);
    acc0 = wmma_bf16(a, b0, acc0);
    acc1 = wmma_bf16(a, b1, acc1);
  }
  const int r = lane & 15, hi = lane >> 4;
#pragma unroll
  for (int i = 0; i < 8; ++i) {
    const int row = wm + i + 8 * hi;
    Y[(size_t)row * N + wn + r]      = acc0[i];
    Y[(size_t)row * N + wn + 16 + r] = acc1[i];
  }
}

// RoPE + (S, nh*HD) -> (nh, S, HD) transpose.
__global__ __launch_bounds__(256) void rope_tr_kernel(const float* __restrict__ in,
                                                      float* __restrict__ out,
                                                      int nheads) {
  const int gid = blockIdx.x * blockDim.x + threadIdx.x;
  const int total = S_LEN * (HD_ / 2);
  if (gid >= total * nheads) return;
  const int j  = gid & 63;
  const int t  = gid >> 6;
  const int hh = t % nheads;
  const int s  = t / nheads;
  const float freq = powf(10000.0f, -(float)(2 * j) / 128.0f);
  const float ang  = (float)s * freq;
  const float c = cosf(ang), sn = sinf(ang);
  const float* src = in + (size_t)s * (nheads * HD_) + hh * HD_;
  const float x1 = src[j];
  const float x2 = src[j + 64];
  float* dst = out + ((size_t)hh * S_LEN + s) * HD_;
  dst[j]      = x1 * c - x2 * sn;
  dst[j + 64] = x2 * c + x1 * sn;
}

// (S, NKV*HD) -> (NKV, S, HD)
__global__ __launch_bounds__(256) void vtr_kernel(const float* __restrict__ in,
                                                  float* __restrict__ out) {
  const int gid = blockIdx.x * blockDim.x + threadIdx.x;
  if (gid >= S_LEN * NKV_ * HD_) return;
  const int d  = gid & 127;
  const int t  = gid >> 7;
  const int hh = t % NKV_;
  const int s  = t / NKV_;
  out[((size_t)hh * S_LEN + s) * HD_ + d] =
      in[(size_t)s * (NKV_ * HD_) + hh * HD_ + d];
}

// Flash-style sliding-window attention.
// Block = 4 waves = one head x 64 queries; waves share double-buffered K/V
// tiles staged with GLOBAL_LOAD_ASYNC_TO_LDS_B128 (ASYNCcnt pipelined against
// the WMMA work of the previous key block). Scores are computed transposed
// (S^T = K * Q^T) so softmax-over-keys reduces in-lane over the 8 C VGPRs,
// finished with one xor-16 shuffle.
__global__ __launch_bounds__(128) void attn_kernel(const float* __restrict__ Qr,   // (NH,S,HD) roped
                                                   const float* __restrict__ Kr,   // (NKV,S,HD) roped
                                                   const float* __restrict__ Vr,   // (NKV,S,HD)
                                                   float* __restrict__ Oout) {     // (S, NH*HD)
  __shared__ __align__(16) float  Klds[2][32 * 128];   // 2 x 16 KB, fp32 key tiles
  __shared__ __align__(16) float  Vlds[2][32 * 128];   // 2 x 16 KB, fp32 value tiles
  __shared__ __align__(16) __bf16 Plds[4][32 * 16];    // per-wave P^T tile [key][query]

  const int h   = blockIdx.x;
  const int q0b = blockIdx.y * 64;
  const int kv  = h >> 2;                 // G = NH/NKV = 4
  const int lane = threadIdx.x & 31;
  const int wv   = threadIdx.x >> 5;      // wave 0..3
  const int q0   = q0b + 16 * wv;         // this wave's 16-query tile
  const int r = lane & 15, hi = lane >> 4;
  const float* Qh = Qr + (size_t)h  * S_LEN * HD_;
  const float* Kh = Kr + (size_t)kv * S_LEN * HD_;
  const float* Vh = Vr + (size_t)kv * S_LEN * HD_;
  const float scale = 0.08838834764831845f;  // 1/sqrt(128)

  // Preload Q tile as four B-fragments (column = query, K = d chunk of 32)
  v16bf qf[4];
#pragma unroll
  for (int c = 0; c < 4; ++c) qf[c] = load_b_frag(Qh, HD_, q0, c * 32, lane);

  v8f o[8] = {};                 // O accumulators: 16 queries x 128 d
  float m_run = -1e30f, l_run = 0.0f;

  // Block-level key range (union over the 4 waves); per-wave active window.
  int jlo_blk = q0b - WIN_; if (jlo_blk < 0) jlo_blk = 0; jlo_blk &= ~31;
  const int jhi_blk = q0b + 48;
  int jlo_w = q0 - WIN_; if (jlo_w < 0) jlo_w = 0; jlo_w &= ~31;

  // Cooperative async stage of one 32-key K+V tile pair: each wave issues
  // 8 K rows + 8 V rows, one 16 B/lane async op per row (512 B per issue).
  auto issue_async = [&](int buf, int jj) {
#pragma unroll
    for (int rr = 0; rr < 8; ++rr) {
      const int row = wv * 8 + rr;
      const unsigned lK =
          (unsigned)(uintptr_t)&Klds[buf][0] + (unsigned)(row * 512 + lane * 16);
      const unsigned lV =
          (unsigned)(uintptr_t)&Vlds[buf][0] + (unsigned)(row * 512 + lane * 16);
      const float* pK = Kh + (size_t)(jj + row) * HD_ + lane * 4;
      const float* pV = Vh + (size_t)(jj + row) * HD_ + lane * 4;
      asm volatile("global_load_async_to_lds_b128 %0, %1, off"
                   :: "v"(lK), "v"(pK) : "memory");
      asm volatile("global_load_async_to_lds_b128 %0, %1, off"
                   :: "v"(lV), "v"(pV) : "memory");
    }
  };

  issue_async(0, jlo_blk);
  int buf = 0;
  for (int j0 = jlo_blk; j0 <= jhi_blk; j0 += 32, buf ^= 1) {
    const bool have_next = (j0 + 32 <= jhi_blk);
    if (have_next) {
      issue_async(buf ^ 1, j0 + 32);  // overlap next tile copy with this compute
      asm volatile("s_wait_asynccnt 0x10" ::: "memory");  // 16 newest still in flight
    } else {
      asm volatile("s_wait_asynccnt 0x0" ::: "memory");
    }
    __syncthreads();

    const bool active = (j0 >= jlo_w) && (j0 <= q0);
    if (active) {
      // ---- scores: St[t] = K(16x128) * Q^T(128x16), key subtile t ----
      v8f st[2] = {};
#pragma unroll
      for (int t = 0; t < 2; ++t)
#pragma unroll
        for (int c = 0; c < 4; ++c) {
          v16bf kf = load_a_frag_lds(&Klds[buf][(16 * t + r) * 128 + c * 32 + hi * 8]);
          st[t] = wmma_bf16(kf, qf[c], st[t]);
        }
      // ---- scale + causal/window mask, per-column (query) max ----
      const int qi = q0 + r;
      float tmax = -1e30f;
#pragma unroll
      for (int t = 0; t < 2; ++t)
#pragma unroll
        for (int i = 0; i < 8; ++i) {
          const int j = j0 + 16 * t + i + 8 * hi;
          const bool ok = (qi >= j) && (qi - j <= WIN_);
          const float v = ok ? st[t][i] * scale : -1e30f;
          st[t][i] = v;
          tmax = fmaxf(tmax, v);
        }
      tmax = fmaxf(tmax, __shfl_xor(tmax, 16, 32));
      const float mnew  = fmaxf(m_run, tmax);
      const float alpha = __expf(m_run - mnew);
      float lsum = 0.0f;
#pragma unroll
      for (int t = 0; t < 2; ++t)
#pragma unroll
        for (int i = 0; i < 8; ++i) {
          const float p = __expf(st[t][i] - mnew);
          st[t][i] = p;
          lsum += p;
        }
      lsum += __shfl_xor(lsum, 16, 32);
      l_run = l_run * alpha + lsum;
      m_run = mnew;
      // rescale O rows by per-query alpha (row q = i + 8*hi; lane q holds alpha_q)
      float ar[8];
#pragma unroll
      for (int i = 0; i < 8; ++i) ar[i] = __shfl(alpha, i + 8 * hi, 32);
#pragma unroll
      for (int n = 0; n < 8; ++n)
#pragma unroll
        for (int i = 0; i < 8; ++i) o[n][i] *= ar[i];

      // ---- P^T -> wave-private LDS (bf16); same-wave LDS ops stay in order ----
#pragma unroll
      for (int t = 0; t < 2; ++t)
#pragma unroll
        for (int i = 0; i < 8; ++i) {
          const int kk = 16 * t + i + 8 * hi;
          Plds[wv][kk * 16 + r] = (__bf16)st[t][i];
        }
      // ---- P as A-fragment (row = query, K = 32 keys) ----
      v16bf pf;
#pragma unroll
      for (int e = 0; e < 8; ++e) pf[e]     = Plds[wv][(hi * 8 + e) * 16 + r];
#pragma unroll
      for (int e = 0; e < 8; ++e) pf[8 + e] = Plds[wv][(16 + hi * 8 + e) * 16 + r];
      // ---- O += P(16x32) * V(32x16) over 8 d-chunks ----
#pragma unroll
      for (int n = 0; n < 8; ++n) {
        v16bf vf;
        const int d = n * 16 + r;
        float t16[16];
#pragma unroll
        for (int e = 0; e < 16; ++e) t16[e] = Vlds[buf][(hi * 8 + (e & 7) + (e >> 3) * 16) * 128 + d];
        vf = pack16(t16);
        o[n] = wmma_bf16(pf, vf, o[n]);
      }
    }
    __syncthreads();  // all waves done with buf before it is refilled
  }

  // ---- normalize and store (row q = i + 8*hi; lane q holds l_q) ----
  float lr[8];
#pragma unroll
  for (int i = 0; i < 8; ++i) lr[i] = __shfl(l_run, i + 8 * hi, 32);
#pragma unroll
  for (int n = 0; n < 8; ++n)
#pragma unroll
    for (int i = 0; i < 8; ++i) {
      const int row = q0 + i + 8 * hi;
      Oout[(size_t)row * (NH_ * HD_) + h * HD_ + n * 16 + r] = o[n][i] / lr[i];
    }
}

extern "C" void kernel_launch(void* const* d_in, const int* in_sizes, int n_in,
                              void* d_out, int out_size, void* d_ws, size_t ws_size,
                              hipStream_t stream) {
  (void)in_sizes; (void)n_in; (void)out_size; (void)ws_size;
  const float* x  = (const float*)d_in[0];
  const float* Wq = (const float*)d_in[1];
  const float* Wk = (const float*)d_in[2];
  const float* Wv = (const float*)d_in[3];
  const float* Wo = (const float*)d_in[4];
  float* out = (float*)d_out;

  // workspace layout (floats); total ~84 MB
  float* ws     = (float*)d_ws;
  float* q_proj = ws;                                    // S * 4096
  float* q_rope = q_proj + (size_t)S_LEN * HID_;         // NH * S * HD
  float* k_proj = q_rope + (size_t)S_LEN * HID_;         // S * 1024
  float* v_proj = k_proj + (size_t)S_LEN * NKV_ * HD_;   // S * 1024
  float* attn   = q_proj;                                // reuse after rope

  float* k_out = out + (size_t)S_LEN * NH_ * HD_;        // (NKV,S,HD) roped K
  float* v_out = k_out + (size_t)NKV_ * S_LEN * HD_;     // (NKV,S,HD) V

  const dim3 blk(256);
  // QKV projections (bf16 WMMA, fp32 accumulate)
  proj_nt_kernel<<<dim3(S_LEN / 64, HID_ / 64), blk, 0, stream>>>(x, Wq, q_proj, HID_, HID_);
  proj_nt_kernel<<<dim3(S_LEN / 64, (NKV_ * HD_) / 64), blk, 0, stream>>>(x, Wk, k_proj, NKV_ * HD_, HID_);
  proj_nt_kernel<<<dim3(S_LEN / 64, (NKV_ * HD_) / 64), blk, 0, stream>>>(x, Wv, v_proj, NKV_ * HD_, HID_);
  // RoPE + head-major transpose (K and V go straight into d_out; they are outputs)
  const int nq = S_LEN * NH_ * (HD_ / 2);
  rope_tr_kernel<<<(nq + 255) / 256, blk, 0, stream>>>(q_proj, q_rope, NH_);
  const int nk = S_LEN * NKV_ * (HD_ / 2);
  rope_tr_kernel<<<(nk + 255) / 256, blk, 0, stream>>>(k_proj, k_out, NKV_);
  const int nv = S_LEN * NKV_ * HD_;
  vtr_kernel<<<(nv + 255) / 256, blk, 0, stream>>>(v_proj, v_out);
  // sliding-window flash attention (4 waves/block, async double-buffered K/V)
  attn_kernel<<<dim3(NH_, S_LEN / 64), dim3(128), 0, stream>>>(q_rope, k_out, v_out, attn);
  // output projection
  proj_nt_kernel<<<dim3(S_LEN / 64, HID_ / 64), blk, 0, stream>>>(attn, Wo, out, HID_, HID_);
}
